// AnchorPlusOffset_77988016161040
// MI455X (gfx1250) — matrix operand order repaired and checked by reading, not compiled, not run on hardware
//
#include <hip/hip_runtime.h>
#include <math.h>

typedef __attribute__((ext_vector_type(16))) __bf16 v16bf;
typedef __attribute__((ext_vector_type(8)))  float  v8f;

#define D_DIM   2048
#define V_DIM   32000
#define BS_DIM  8192
#define MB      64            // x-rows per workgroup in GEMM
#define NPASS   125           // col passes per V-slice: 125 * 128 = 16000
#define VSLICE  16000         // V split in 2 slices across blockIdx.y

__device__ __forceinline__ unsigned int f2bf_pair(float lo, float hi) {
    unsigned int a = __float_as_uint(lo);
    unsigned int b = __float_as_uint(hi);
    a += 0x7FFFu + ((a >> 16) & 1u);   // round-to-nearest-even
    b += 0x7FFFu + ((b >> 16) & 1u);
    return (a >> 16) | (b & 0xFFFF0000u);
}

// ---------------- Kernel 1: convert x (unnormalized) to bf16 -------------
__global__ void convert_x_bf16(const float* __restrict__ x,
                               unsigned short* __restrict__ xb) {
    size_t i = (size_t)blockIdx.x * blockDim.x + threadIdx.x;   // 8 elems/thread
    const float4* xp = (const float4*)x + i * 2;
    float4 a = xp[0];
    float4 b = xp[1];
    uint4 o;
    o.x = f2bf_pair(a.x, a.y);
    o.y = f2bf_pair(a.z, a.w);
    o.z = f2bf_pair(b.x, b.y);
    o.w = f2bf_pair(b.z, b.w);
    *(uint4*)(xb + i * 8) = o;
}

// ---------------- Kernel 2: normalize vocab rows, convert to bf16 --------
__global__ void normalize_vocab_bf16(const float* __restrict__ v,
                                     unsigned short* __restrict__ vb) {
    __shared__ float red[256];
    const int row = blockIdx.x;
    const int tid = threadIdx.x;
    const float4* vr = (const float4*)(v + (size_t)row * D_DIM);
    float4 c0 = vr[tid * 2];
    float4 c1 = vr[tid * 2 + 1];
    float ss = c0.x*c0.x + c0.y*c0.y + c0.z*c0.z + c0.w*c0.w
             + c1.x*c1.x + c1.y*c1.y + c1.z*c1.z + c1.w*c1.w;
    red[tid] = ss;
    __syncthreads();
    for (int s = 128; s > 0; s >>= 1) {
        if (tid < s) red[tid] += red[tid + s];
        __syncthreads();
    }
    float inv = 1.0f / fmaxf(sqrtf(red[0]), 1e-12f);
    uint4 o;
    o.x = f2bf_pair(c0.x * inv, c0.y * inv);
    o.y = f2bf_pair(c0.z * inv, c0.w * inv);
    o.z = f2bf_pair(c1.x * inv, c1.y * inv);
    o.w = f2bf_pair(c1.z * inv, c1.w * inv);
    *(uint4*)(vb + (size_t)row * D_DIM + tid * 8) = o;
}

// ---------------- Kernel 3: bf16 WMMA GEMM + per-row argmax --------------
// LDS layout: fragment-ordered. For each (kc, strip) a 1KB tile of 64 int4:
//   slot [lane]      = lane's A-fragment K-octets {lhalf*8+0..7}      (VGPR0-3)
//   slot [32 + lane] = lane's A-fragment K-octets {16+lhalf*8+0..7}   (VGPR4-7)
// All strip/half offsets are small -> DS immediate offsets, no VALU adds.
union Frag { int4 i[2]; v16bf v; };

__global__ __launch_bounds__(256)
void gemm_argmax_kernel(const unsigned short* __restrict__ xb,
                        const unsigned short* __restrict__ vb,
                        float* __restrict__ bVal, int* __restrict__ bIdx) {
    __shared__ int4 sA4[64 * 4 * 64];     // 64 kc-chunks * 4 strips * 1KB = 256KB
    const int tid    = threadIdx.x;
    const int mblk   = blockIdx.x;
    const int vslice = blockIdx.y;
    const int wave   = tid >> 5;
    const int lane   = tid & 31;
    const int l15    = lane & 15;
    const int lhalf  = lane >> 4;

    // Stage the 64-row x block into LDS, permuted into fragment order.
    const unsigned short* xrow = xb + (size_t)mblk * MB * D_DIM;
    for (int c = tid; c < MB * 256; c += 256) {
        int r  = c >> 8;                // row in block (0..63)
        int o  = c & 255;               // 8-element octet index within row
        int kc = o >> 2;
        int j  = o & 3;                 // octet within 32-K chunk
        int s  = r >> 4;
        int m  = r & 15;
        int4 t = *(const int4*)(xrow + (size_t)r * D_DIM + o * 8);
        //        tile base        | half       | lane slot
        sA4[(kc * 4 + s) * 64 + (j >> 1) * 32 + (j & 1) * 16 + m] = t;
    }
    __syncthreads();

    float bval[4][8];
    int   bidx[4][8];
#pragma unroll
    for (int s = 0; s < 4; ++s)
#pragma unroll
        for (int r = 0; r < 8; ++r) { bval[s][r] = -__builtin_inff(); bidx[s][r] = 0; }

    const int colWave = vslice * VSLICE + wave * 16;
    for (int pass = 0; pass < NPASS; ++pass) {
        const int col = colWave + (pass << 7) + l15;     // this lane's B column
        const unsigned short* bp = vb + (size_t)col * D_DIM + lhalf * 8;
        v8f acc[4] = {};
#pragma unroll 4
        for (int kc = 0; kc < 64; ++kc) {
            Frag B;                                       // B 32x16 bf16 fragment
            B.i[0] = *(const int4*)(bp);
            B.i[1] = *(const int4*)(bp + 16);
            bp += 32;
            const int4* tb = &sA4[kc * 256 + lane];       // kc*4 tiles * 64 int4
#pragma unroll
            for (int s = 0; s < 4; ++s) {                 // 4 row strips of 16
                Frag A;                                   // A 16x32 bf16 fragment
                A.i[0] = tb[s * 64];                      // +s*1KB      (imm)
                A.i[1] = tb[s * 64 + 32];                 // +s*1KB+512  (imm)
                acc[s] = __builtin_amdgcn_wmma_f32_16x16x32_bf16(
                    false, A.v, false, B.v, (short)0, acc[s], false, false);
            }
        }
        // per-lane running argmax (no cross-lane traffic in hot loop)
#pragma unroll
        for (int s = 0; s < 4; ++s)
#pragma unroll
            for (int r = 0; r < 8; ++r) {
                float v = acc[s][r];
                if (v > bval[s][r]) { bval[s][r] = v; bidx[s][r] = col; }
            }
    }

    // Butterfly reduce across the 16 lanes holding one C row (N dimension)
#pragma unroll
    for (int s = 0; s < 4; ++s)
#pragma unroll
        for (int r = 0; r < 8; ++r) {
            float v = bval[s][r];
            int  id = bidx[s][r];
#pragma unroll
            for (int m = 1; m < 16; m <<= 1) {
                float ov = __shfl_xor(v, m, 32);
                int   oi = __shfl_xor(id, m, 32);
                if (ov > v || (ov == v && oi < id)) { v = ov; id = oi; }
            }
            if (l15 == 0) {
                int grow = mblk * MB + s * 16 + 8 * lhalf + r;  // C row mapping
                bVal[grow * 2 + vslice] = v;
                bIdx[grow * 2 + vslice] = id;
            }
        }
}

// ---------------- Kernel 4: gather anchor, scaled offset, outputs --------
__global__ void finalize_kernel(const float* __restrict__ x,
                                const float* __restrict__ vocab,
                                const float* __restrict__ bVal,
                                const int* __restrict__ bIdx,
                                float* __restrict__ outRes,
                                float* __restrict__ outIds) {
    __shared__ float redA[256];
    __shared__ float redO[256];
    const int row = blockIdx.x;
    const int tid = threadIdx.x;

    float v0 = bVal[row * 2 + 0], v1 = bVal[row * 2 + 1];
    int   i0 = bIdx[row * 2 + 0], i1 = bIdx[row * 2 + 1];
    int id = (v1 > v0 || (v1 == v0 && i1 < i0)) ? i1 : i0;

    const float4* xr = (const float4*)(x + (size_t)row * D_DIM);
    const float4* ar = (const float4*)(vocab + (size_t)id * D_DIM);
    float4 xa = xr[tid * 2], xb4 = xr[tid * 2 + 1];
    float4 aa = ar[tid * 2], ab4 = ar[tid * 2 + 1];

    float sa = aa.x*aa.x + aa.y*aa.y + aa.z*aa.z + aa.w*aa.w
             + ab4.x*ab4.x + ab4.y*ab4.y + ab4.z*ab4.z + ab4.w*ab4.w;
    float ox0 = xa.x - aa.x, ox1 = xa.y - aa.y, ox2 = xa.z - aa.z, ox3 = xa.w - aa.w;
    float ox4 = xb4.x - ab4.x, ox5 = xb4.y - ab4.y, ox6 = xb4.z - ab4.z, ox7 = xb4.w - ab4.w;
    float so = ox0*ox0 + ox1*ox1 + ox2*ox2 + ox3*ox3
             + ox4*ox4 + ox5*ox5 + ox6*ox6 + ox7*ox7;

    redA[tid] = sa; redO[tid] = so;
    __syncthreads();
    for (int s = 128; s > 0; s >>= 1) {
        if (tid < s) { redA[tid] += redA[tid + s]; redO[tid] += redO[tid + s]; }
        __syncthreads();
    }
    float anorm = sqrtf(redA[0]);
    float onorm = sqrtf(redO[0]);
    float scale = fminf(0.1f * anorm / (onorm + 1e-8f), 1.0f);

    float4 r0, r1;
    r0.x = aa.x + ox0 * scale;  r0.y = aa.y + ox1 * scale;
    r0.z = aa.z + ox2 * scale;  r0.w = aa.w + ox3 * scale;
    r1.x = ab4.x + ox4 * scale; r1.y = ab4.y + ox5 * scale;
    r1.z = ab4.z + ox6 * scale; r1.w = ab4.w + ox7 * scale;
    float4* orow = (float4*)(outRes + (size_t)row * D_DIM);
    orow[tid * 2]     = r0;
    orow[tid * 2 + 1] = r1;
    if (tid == 0) outIds[row] = (float)id;
}

extern "C" void kernel_launch(void* const* d_in, const int* in_sizes, int n_in,
                              void* d_out, int out_size, void* d_ws, size_t ws_size,
                              hipStream_t stream) {
    const float* x     = (const float*)d_in[0];   // [4,2048,2048] fp32
    const float* vocab = (const float*)d_in[1];   // [32000,2048] fp32

    char* ws = (char*)d_ws;
    unsigned short* xb = (unsigned short*)ws;                                  // 32 MB
    unsigned short* vb = (unsigned short*)(ws + (size_t)BS_DIM * D_DIM * 2);   // 128 MB
    char* tail = ws + (size_t)BS_DIM * D_DIM * 2 + (size_t)V_DIM * D_DIM * 2;
    float* bVal = (float*)tail;                         // [8192][2]
    int*   bIdx = (int*)(tail + BS_DIM * 2 * sizeof(float));

    float* outRes = (float*)d_out;
    float* outIds = outRes + (size_t)BS_DIM * D_DIM;

    convert_x_bf16<<<(BS_DIM * D_DIM / 8) / 256, 256, 0, stream>>>(x, xb);
    normalize_vocab_bf16<<<V_DIM, 256, 0, stream>>>(vocab, vb);
    gemm_argmax_kernel<<<dim3(BS_DIM / MB, 2), 256, 0, stream>>>(xb, vb, bVal, bIdx);
    finalize_kernel<<<BS_DIM, 256, 0, stream>>>(x, vocab, bVal, bIdx, outRes, outIds);
}
